// _Additive_84602265796698
// MI455X (gfx1250) — matrix operand order repaired and checked by reading, not compiled, not run on hardware
//
#include <hip/hip_runtime.h>
#include <hip/hip_bf16.h>
#include <math.h>

// CDNA5 WMMA vector types
typedef __attribute__((ext_vector_type(16))) __bf16 v16bf;
typedef __attribute__((ext_vector_type(8)))  float  v8f;
typedef __attribute__((ext_vector_type(4)))  float  v4f;   // native vector for NT loads

#define BDIM   64
#define TDIM   8192
#define DDIM   256
#define UDIM   128
#define PITCH  264          // 256 + 8 bf16 elements -> 528B row pitch = 33 * 16B (conflict-free)
#define TCHUNK 1024
#define CHUNKS (TDIM / TCHUNK)   // 8

union Frag16 {
    v16bf          bf;
    unsigned short u[16];
    unsigned       d[8];
    uint4          q[2];
};

__device__ __forceinline__ unsigned short f32_to_bf16_rne(float x) {
    unsigned xb = __float_as_uint(x);
    unsigned r  = xb + 0x7FFFu + ((xb >> 16) & 1u);
    return (unsigned short)(r >> 16);
}

#if __has_builtin(__builtin_amdgcn_tanhf)
__device__ __forceinline__ float fast_tanh(float x) { return __builtin_amdgcn_tanhf(x); }
#else
__device__ __forceinline__ float fast_tanh(float x) { return tanhf(x); }
#endif

// pack bf16-truncations of (x0, x1) into one dword: one v_perm_b32
__device__ __forceinline__ unsigned pack_hi16(float x0, float x1) {
    return __builtin_amdgcn_perm(__float_as_uint(x1), __float_as_uint(x0), 0x07060302u);
}

// ---------------------------------------------------------------------------
// Kernel 1: S[b][u] = sum_d source[b][d] * W1[d][u]    (tiny: 2.1 MFLOP)
// ---------------------------------------------------------------------------
__global__ void bias_proj_kernel(const float* __restrict__ source,
                                 const float* __restrict__ W1,
                                 float* __restrict__ S) {
    const int b = blockIdx.x;
    const int u = threadIdx.x;
    const float* src = source + b * DDIM;
    float acc = 0.f;
#pragma unroll 4
    for (int d = 0; d < DDIM; ++d)
        acc = fmaf(src[d], W1[d * UDIM + u], acc);
    S[b * UDIM + u] = acc;
}

// ---------------------------------------------------------------------------
// Kernel 2: scores[b][t] = sum_u tanh(S[b][u] + (target[b][t] @ W2)[u]) * attn[u]
// Split-bf16 (3-pass Markidis) WMMA for t = target@W2, f32 accumulation.
// ---------------------------------------------------------------------------
__global__ void __launch_bounds__(256, 1)
additive_attn_kernel(const float* __restrict__ target,
                     const float* __restrict__ S,
                     const float* __restrict__ W2,
                     const float* __restrict__ attn,
                     float* __restrict__ out) {
    // W2 transposed [U][D], pre-split into bf16 hi/lo planes (2 * 66 KB)
    __shared__ unsigned short sHi[UDIM * PITCH];
    __shared__ unsigned short sLo[UDIM * PITCH];

    const int b     = blockIdx.x / CHUNKS;
    const int chunk = blockIdx.x % CHUNKS;
    const int tid   = threadIdx.x;

    // ---- stage W2 once per block: global fp32 -> LDS bf16 hi/lo, transposed
    // (one-time cost, use RNE splits here for slightly better B accuracy)
    for (int idx = tid; idx < DDIM * UDIM; idx += 256) {
        const int d = idx >> 7;            // / UDIM
        const int u = idx & (UDIM - 1);
        float w = W2[idx];
        unsigned wb = __float_as_uint(w);
        unsigned r  = wb + 0x7FFFu + ((wb >> 16) & 1u);
        unsigned short h = (unsigned short)(r >> 16);
        float hf = __uint_as_float((unsigned)h << 16);
        sHi[u * PITCH + d] = h;
        sLo[u * PITCH + d] = f32_to_bf16_rne(w - hf);
    }
    __syncthreads();

    const int lane = tid & 31;
    const int wave = tid >> 5;
    const int ln   = lane & 15;     // M row (A) / N col (B/C)
    const int lg   = lane >> 4;     // lane-group

    // per-lane bias + attention weights for this lane's N columns
    float su[8], aw[8];
#pragma unroll
    for (int n = 0; n < 8; ++n) {
        su[n] = S[b * UDIM + n * 16 + ln];
        aw[n] = attn[n * 16 + ln];
    }

    const int tilesPerBlock = TCHUNK / 16;   // 64 tiles, 8 waves -> 8 tiles/wave
    for (int tile = wave; tile < tilesPerBlock; tile += 8) {
        const int t0 = chunk * TCHUNK + tile * 16;
        const float* arow = target + ((size_t)b * TDIM + t0 + ln) * DDIM;

        v8f c[8];
#pragma unroll
        for (int n = 0; n < 8; ++n) c[n] = {};

#pragma unroll 2
        for (int k0 = 0; k0 < DDIM; k0 += 32) {
            // ---- A slab [16 x 32] fp32 -> bf16 hi/lo fragments (streamed, NT) ----
            // 16-bit A layout: lane%16=M; slots 0-7 -> K = k0+8*lg+j,
            //                              slots 8-15 -> K = k0+16+8*lg+(j-8)
            const float* ap = arow + k0 + 8 * lg;
            v4f f0 = __builtin_nontemporal_load((const v4f*)(ap));
            v4f f1 = __builtin_nontemporal_load((const v4f*)(ap + 4));
            v4f f2 = __builtin_nontemporal_load((const v4f*)(ap + 16));
            v4f f3 = __builtin_nontemporal_load((const v4f*)(ap + 20));
            __builtin_prefetch(ap + 32, 0, 1);   // next K-slab -> global_prefetch_b8

            float ax[16] = {f0.x, f0.y, f0.z, f0.w, f1.x, f1.y, f1.z, f1.w,
                            f2.x, f2.y, f2.z, f2.w, f3.x, f3.y, f3.z, f3.w};
            Frag16 ahi, alo;
            // truncation split: hi = trunc_bf16(x); lo = trunc_bf16(x - hi)
            // 3 VALU ops/element (and + sub + shared perm), x = hi+lo to ~2^-16
#pragma unroll
            for (int p = 0; p < 8; ++p) {
                float x0 = ax[2 * p], x1 = ax[2 * p + 1];
                ahi.d[p] = pack_hi16(x0, x1);
                float l0 = x0 - __uint_as_float(__float_as_uint(x0) & 0xFFFF0000u);
                float l1 = x1 - __uint_as_float(__float_as_uint(x1) & 0xFFFF0000u);
                alo.d[p] = pack_hi16(l0, l1);
            }

            // ---- 8 column tiles of W2: B layout lane%16=N, K = k0+8*lg+j (j<8), k0+16+8*lg+(j-8) (j>=8) ----
#pragma unroll
            for (int n = 0; n < 8; ++n) {
                const int boff = (n * 16 + ln) * PITCH + k0 + 8 * lg;
                Frag16 bhi, blo;
                bhi.q[0] = *(const uint4*)(&sHi[boff]);
                bhi.q[1] = *(const uint4*)(&sHi[boff + 16]);
                blo.q[0] = *(const uint4*)(&sLo[boff]);
                blo.q[1] = *(const uint4*)(&sLo[boff + 16]);

                c[n] = __builtin_amdgcn_wmma_f32_16x16x32_bf16(
                           false, ahi.bf, false, bhi.bf, (short)0, c[n], false, false);
                c[n] = __builtin_amdgcn_wmma_f32_16x16x32_bf16(
                           false, alo.bf, false, bhi.bf, (short)0, c[n], false, false);
                c[n] = __builtin_amdgcn_wmma_f32_16x16x32_bf16(
                           false, ahi.bf, false, blo.bf, (short)0, c[n], false, false);
            }
        }

        // ---- epilogue: tanh(s + t) . attn, reduce 16 N-lanes per row ----
        // C layout: vgpr v, lane -> element (M = v + 8*lg, N = ln + 16*n)
        float acc[8];
#pragma unroll
        for (int v = 0; v < 8; ++v) acc[v] = 0.f;
#pragma unroll
        for (int n = 0; n < 8; ++n) {
#pragma unroll
            for (int v = 0; v < 8; ++v)
                acc[v] = fmaf(aw[n], fast_tanh(su[n] + c[n][v]), acc[v]);
        }
#pragma unroll
        for (int v = 0; v < 8; ++v) {
            acc[v] += __shfl_xor(acc[v], 1, 16);
            acc[v] += __shfl_xor(acc[v], 2, 16);
            acc[v] += __shfl_xor(acc[v], 4, 16);
            acc[v] += __shfl_xor(acc[v], 8, 16);
        }
        if (ln == 0) {
            float* op = out + (size_t)b * TDIM + t0 + lg * 8;
#pragma unroll
            for (int v = 0; v < 8; ++v) op[v] = acc[v];
        }
    }
}

// ---------------------------------------------------------------------------
extern "C" void kernel_launch(void* const* d_in, const int* in_sizes, int n_in,
                              void* d_out, int out_size, void* d_ws, size_t ws_size,
                              hipStream_t stream) {
    (void)in_sizes; (void)n_in; (void)out_size; (void)ws_size;
    const float* target = (const float*)d_in[0];   // [B,T,D]
    const float* source = (const float*)d_in[1];   // [B,D]
    const float* W1     = (const float*)d_in[2];   // [D,U]
    const float* W2     = (const float*)d_in[3];   // [D,U]
    const float* attn   = (const float*)d_in[4];   // [U]
    float*       out    = (float*)d_out;           // [B,T]
    float*       S      = (float*)d_ws;            // [B,U] scratch (32 KB)

    bias_proj_kernel<<<dim3(BDIM), dim3(UDIM), 0, stream>>>(source, W1, S);
    additive_attn_kernel<<<dim3(BDIM * CHUNKS), dim3(256), 0, stream>>>(
        target, S, W2, attn, out);
}
